// Transformer_26259430048161
// MI455X (gfx1250) — matrix operand order repaired and checked by reading, not compiled, not run on hardware
//
#include <hip/hip_runtime.h>

#define NN   50000
#define EE   800000
#define DD   128
#define HH   8
#define DFFN 512
#define LL   3

typedef __attribute__((ext_vector_type(16))) _Float16     v16h;
typedef __attribute__((ext_vector_type(8)))  float        v8f;
typedef __attribute__((ext_vector_type(4)))  _Float16     h4;
typedef __attribute__((ext_vector_type(4)))  unsigned int u32x4;
typedef __attribute__((ext_vector_type(8)))  int          i32x8;
typedef __attribute__((ext_vector_type(4)))  int          i32x4;

// ---------------------------------------------------------------------------
// TDM: stage a contiguous 2D f16 tile [rows x K] (row stride = K elements)
// from global memory into LDS at byte offset lds_off.  D# per CDNA5 ISA §8:
//   group0: count=1 | lds_addr | global_addr(57b) | type=2
//   group1: data_size=1(2B), tensor_dim0=K, tensor_dim1=tot_rows,
//           tile_dim0=K, tile_dim1=rows, tensor_dim0_stride=K
// Tracked with TENSORcnt -> s_wait_tensorcnt 0.
// ---------------------------------------------------------------------------
__device__ __forceinline__ void tdm_load_tile_f16(unsigned int lds_off,
                                                  const _Float16* gsrc,
                                                  int K, int rows, int tot_rows)
{
    unsigned long long ga = (unsigned long long)(size_t)gsrc;
    u32x4 g0;
    g0[0] = 1u;                                        // count=1, user D#
    g0[1] = lds_off;                                   // LDS byte address
    g0[2] = (unsigned int)ga;                          // global_addr[31:0]
    g0[3] = (unsigned int)(ga >> 32) | (2u << 30);     // addr[56:32] | type=2
    i32x8 g1;
    g1[0] = 1 << 16;                                   // data_size = 2 bytes
    g1[1] = (K & 0xFFFF) << 16;                        // tensor_dim0[15:0]
    g1[2] = ((unsigned)K >> 16) | ((tot_rows & 0xFFFF) << 16);
    g1[3] = (((unsigned)tot_rows >> 16) & 0xFFFF) | (K << 16);  // tile_dim0=K
    g1[4] = rows & 0xFFFF;                             // tile_dim1 = rows
    g1[5] = K;                                         // dim0_stride[31:0]
    g1[6] = 0;
    g1[7] = 0;
    i32x4 z4 = {0, 0, 0, 0};
#if defined(__clang_major__) && (__clang_major__ >= 23)
    i32x8 z8 = {0, 0, 0, 0, 0, 0, 0, 0};
    __builtin_amdgcn_tensor_load_to_lds(g0, g1, z4, z4, z8, 0);
#else
    __builtin_amdgcn_tensor_load_to_lds(g0, g1, z4, z4, 0);
#endif
}

// ---------------------------------------------------------------------------
// Tiled WMMA GEMM: C[M,Nout] = A[M,K](f16) @ B[Nout,K](f16)^T + bias, opt ReLU.
// grid.x = M/16; blockDim = (32, Nout/64): each wave owns a 16(M) x 64(N)
// tile -> 4 accumulators, A fragment reused across 4 WMMAs per k-step.
// The block's shared A tile [16 x K] is staged into LDS by the TDM (wave 0),
// synchronized via s_wait_tensorcnt + workgroup barrier; fragments come from
// LDS (ds_read_b128) while B streams from global (L2-resident weights).
// Fragment layouts per CDNA5 ISA 7.12.2.
// ---------------------------------------------------------------------------
__global__ __launch_bounds__(256)
void gt_gemm_wmma(const _Float16* __restrict__ A, const _Float16* __restrict__ B,
                  const float* __restrict__ bias, float* __restrict__ C32,
                  _Float16* __restrict__ C16, int M, int Nout, int K, int relu)
{
    __shared__ __align__(16) _Float16 Atile[16 * 512];   // up to K=512

    int mt   = blockIdx.x;
    int lane = threadIdx.x;
    int l15  = lane & 15;
    int hi   = lane >> 4;                 // lane-half selects K sub-block

    // ---- stage A tile [16 x K] into LDS with the Tensor Data Mover ----
    if (threadIdx.y == 0) {
        tdm_load_tile_f16((unsigned int)(size_t)&Atile[0],
                          A + (size_t)mt * 16 * K, K, 16, M);
        __builtin_amdgcn_s_wait_tensorcnt(0);
    }
    __syncthreads();

    const _Float16* alds = &Atile[(size_t)l15 * K];
    int nb = threadIdx.y * 64;            // this wave's 64-col slab
    const _Float16* brow0 = B + (size_t)(nb + l15) * K;
    const _Float16* brow1 = brow0 + (size_t)16 * K;
    const _Float16* brow2 = brow0 + (size_t)32 * K;
    const _Float16* brow3 = brow0 + (size_t)48 * K;

    v8f acc0 = {}, acc1 = {}, acc2 = {}, acc3 = {};
    for (int k0 = 0; k0 < K; k0 += 32) {
        union { float4 f[2]; v16h v; } ua, ub0, ub1, ub2, ub3;
        // A 16x32 f16 frag: lanes 0-15 K {0..7,16..23}; lanes 16-31 {8..15,24..31}
        ua.f[0] = *(const float4*)(alds + k0 + hi * 8);
        ua.f[1] = *(const float4*)(alds + k0 + 16 + hi * 8);
        // B 32x16 f16 frags: col = lane&15; lanes 0-15 K {0..7,16..23}; lanes 16-31 {8..15,24..31}
        const float4* bp0 = (const float4*)(brow0 + k0 + hi * 8);
        const float4* bp1 = (const float4*)(brow1 + k0 + hi * 8);
        const float4* bp2 = (const float4*)(brow2 + k0 + hi * 8);
        const float4* bp3 = (const float4*)(brow3 + k0 + hi * 8);
        ub0.f[0] = bp0[0]; ub0.f[1] = bp0[2];
        ub1.f[0] = bp1[0]; ub1.f[1] = bp1[2];
        ub2.f[0] = bp2[0]; ub2.f[1] = bp2[2];
        ub3.f[0] = bp3[0]; ub3.f[1] = bp3[2];
        acc0 = __builtin_amdgcn_wmma_f32_16x16x32_f16(false, ua.v, false, ub0.v, (short)0, acc0, false, false);
        acc1 = __builtin_amdgcn_wmma_f32_16x16x32_f16(false, ua.v, false, ub1.v, (short)0, acc1, false, false);
        acc2 = __builtin_amdgcn_wmma_f32_16x16x32_f16(false, ua.v, false, ub2.v, (short)0, acc2, false, false);
        acc3 = __builtin_amdgcn_wmma_f32_16x16x32_f16(false, ua.v, false, ub3.v, (short)0, acc3, false, false);
    }

    int rbase = mt * 16 + hi * 8;         // f32 C frag: vgpr i -> M = i + 8*hi
    v8f accs[4] = {acc0, acc1, acc2, acc3};
#pragma unroll
    for (int t = 0; t < 4; ++t) {
        int   col = nb + t * 16 + l15;
        float bv  = bias ? bias[col] : 0.0f;
#pragma unroll
        for (int i = 0; i < 8; ++i) {
            float val = accs[t][i] + bv;
            if (relu) val = fmaxf(val, 0.0f);
            size_t idx = (size_t)(rbase + i) * Nout + col;
            if (C32) C32[idx] = val;
            if (C16) C16[idx] = (_Float16)val;
        }
    }
}

// ---------------------------------------------------------------------------
// Elementwise helpers
// ---------------------------------------------------------------------------
__global__ void gt_cast_f16(const float* __restrict__ in, _Float16* __restrict__ out, int n)
{
    int i = blockIdx.x * blockDim.x + threadIdx.x;
    if (i < n) out[i] = (_Float16)in[i];
}

// pad K: 16 -> 32 with zeros while casting (for nf and enc_W)
__global__ void gt_padcast_f16(const float* __restrict__ in, _Float16* __restrict__ out, int rows)
{
    int i = blockIdx.x * blockDim.x + threadIdx.x;
    if (i >= rows * 32) return;
    int r = i >> 5, c = i & 31;
    out[i] = (c < 16) ? (_Float16)in[r * 16 + c] : (_Float16)0.0f;
}

__global__ void gt_zero_f32(float* __restrict__ p, int n)
{
    int i = blockIdx.x * blockDim.x + threadIdx.x;
    if (i < n) p[i] = 0.0f;
}

// ---------------------------------------------------------------------------
// Edge attention. Clamp to +-5 => exp is numerically safe without max shift
// (softmax is shift invariant), so single pass: e = exp(u), s = segsum(e).
// One thread per (edge, head); gathers hit the 192MB L2 (node arrays ~26MB).
// ---------------------------------------------------------------------------
__global__ void gt_edge_logits(const float* __restrict__ q, const float* __restrict__ k,
                               const int* __restrict__ src, const int* __restrict__ dst,
                               float* __restrict__ e, float* __restrict__ s)
{
    int idx = blockIdx.x * blockDim.x + threadIdx.x;
    if (idx >= EE * HH) return;
    int ed = idx >> 3, hh = idx & 7;
    int sN = src[ed], dN = dst[ed];
    const float4* qp = (const float4*)(q + (size_t)dN * DD + hh * 16);
    const float4* kp = (const float4*)(k + (size_t)sN * DD + hh * 16);
    float u = 0.0f;
#pragma unroll
    for (int j = 0; j < 4; ++j) {
        float4 a = qp[j], b = kp[j];
        u += a.x * b.x + a.y * b.y + a.z * b.z + a.w * b.w;
    }
    u *= 0.25f;                                  // 1/sqrt(DH)
    u = fminf(5.0f, fmaxf(-5.0f, u));
    float ex = __expf(u);
    e[idx] = ex;
    atomicAdd(&s[(size_t)dN * HH + hh], ex);
}

__global__ void gt_edge_aggregate(const float* __restrict__ v, const float* __restrict__ e,
                                  const float* __restrict__ s, const int* __restrict__ src,
                                  const int* __restrict__ dst, float* __restrict__ agg)
{
    int idx = blockIdx.x * blockDim.x + threadIdx.x;
    if (idx >= EE * HH) return;
    int ed = idx >> 3, hh = idx & 7;
    int sN = src[ed], dN = dst[ed];
    float a = e[idx] / s[(size_t)dN * HH + hh];
    const float* vp = v + (size_t)sN * DD + hh * 16;
    float* op = agg + (size_t)dN * DD + hh * 16;
#pragma unroll
    for (int j = 0; j < 16; ++j) atomicAdd(op + j, a * vp[j]);
}

// ---------------------------------------------------------------------------
// Residual + LayerNorm: out = LN(x + r) * g + b ; writes f32 and f16 copies.
// One wave (32 lanes) per row of 128; shfl_xor tree reductions (wave32).
// ---------------------------------------------------------------------------
__global__ __launch_bounds__(256)
void gt_add_ln(const float* __restrict__ x, const float* __restrict__ r,
               const float* __restrict__ g, const float* __restrict__ b,
               float* __restrict__ o32, _Float16* __restrict__ o16, int rows)
{
    int row = blockIdx.x * 8 + threadIdx.y;
    if (row >= rows) return;
    int lane = threadIdx.x;
    float4 xv = ((const float4*)(x + (size_t)row * DD))[lane];
    float4 rv = ((const float4*)(r + (size_t)row * DD))[lane];
    float v0 = xv.x + rv.x, v1 = xv.y + rv.y, v2 = xv.z + rv.z, v3 = xv.w + rv.w;

    float sum = v0 + v1 + v2 + v3;
#pragma unroll
    for (int m = 16; m >= 1; m >>= 1) sum += __shfl_xor(sum, m, 32);
    float mu = sum * (1.0f / 128.0f);

    float d0 = v0 - mu, d1 = v1 - mu, d2 = v2 - mu, d3 = v3 - mu;
    float qv = d0 * d0 + d1 * d1 + d2 * d2 + d3 * d3;
#pragma unroll
    for (int m = 16; m >= 1; m >>= 1) qv += __shfl_xor(qv, m, 32);
    float rstd = rsqrtf(qv * (1.0f / 128.0f) + 1e-5f);

    float4 gv = ((const float4*)g)[lane];
    float4 bv = ((const float4*)b)[lane];
    float o0 = d0 * rstd * gv.x + bv.x;
    float o1 = d1 * rstd * gv.y + bv.y;
    float o2 = d2 * rstd * gv.z + bv.z;
    float o3 = d3 * rstd * gv.w + bv.w;

    float4 ov; ov.x = o0; ov.y = o1; ov.z = o2; ov.w = o3;
    ((float4*)(o32 + (size_t)row * DD))[lane] = ov;
    h4 hv; hv.x = (_Float16)o0; hv.y = (_Float16)o1; hv.z = (_Float16)o2; hv.w = (_Float16)o3;
    ((h4*)(o16 + (size_t)row * DD))[lane] = hv;
}

// ---------------------------------------------------------------------------
// Decoder + mean readout
// ---------------------------------------------------------------------------
__global__ void gt_out_init(float* __restrict__ out, const float* __restrict__ decb)
{
    if (blockIdx.x == 0 && threadIdx.x == 0) out[0] = decb[0];
}

__global__ void gt_decode_mean(const float* __restrict__ h, const float* __restrict__ w,
                               float* __restrict__ out)
{
    int row = blockIdx.x * blockDim.x + threadIdx.x;
    if (row >= NN) return;
    const float4* hp = (const float4*)(h + (size_t)row * DD);
    const float4* wp = (const float4*)w;
    float acc = 0.0f;
#pragma unroll
    for (int j = 0; j < 32; ++j) {
        float4 a = hp[j], c = wp[j];
        acc += a.x * c.x + a.y * c.y + a.z * c.z + a.w * c.w;
    }
    atomicAdd(out, acc * (1.0f / (float)NN));
}

// ---------------------------------------------------------------------------
// Host orchestration
// ---------------------------------------------------------------------------
extern "C" void kernel_launch(void* const* d_in, const int* in_sizes, int n_in,
                              void* d_out, int out_size, void* d_ws, size_t ws_size,
                              hipStream_t stream)
{
    const float* nf    = (const float*)d_in[0];
    const int*   src   = (const int*)d_in[1];
    const int*   dst   = (const int*)d_in[2];
    const float* enc_W = (const float*)d_in[3];
    const float* enc_b = (const float*)d_in[4];
    const float* Wq    = (const float*)d_in[5];
    const float* Wk    = (const float*)d_in[6];
    const float* Wv    = (const float*)d_in[7];
    const float* Wo    = (const float*)d_in[8];
    const float* ln1g  = (const float*)d_in[9];
    const float* ln1b  = (const float*)d_in[10];
    const float* W1    = (const float*)d_in[11];
    const float* b1    = (const float*)d_in[12];
    const float* W2    = (const float*)d_in[13];
    const float* b2    = (const float*)d_in[14];
    const float* ln2g  = (const float*)d_in[15];
    const float* ln2b  = (const float*)d_in[16];
    const float* decW  = (const float*)d_in[17];
    const float* decb  = (const float*)d_in[18];
    float* out = (float*)d_out;

    // ---- workspace layout (bytes), with lifetime-based reuse ----
    const size_t ND4  = (size_t)NN * DD * 4;      // 25.6 MB
    const size_t ND2  = (size_t)NN * DD * 2;      // 12.8 MB
    const size_t NF2  = (size_t)NN * DFFN * 2;    // 51.2 MB
    const size_t EH4  = (size_t)EE * HH * 4;      // 25.6 MB
    const size_t NH4  = (size_t)NN * HH * 4;      //  1.6 MB

    char* ws = (char*)d_ws;
    size_t off = 0;
    float*    h32   = (float*)(ws + off);    off += ND4;
    _Float16* h16   = (_Float16*)(ws + off); off += ND2;
    char*     bufA  = ws + off;              off += NF2;   // q | k, later mid16
    float*    bufB  = (float*)(ws + off);    off += ND4;   // v, later f
    float*    bufC  = (float*)(ws + off);    off += EH4;   // e, later uh
    float*    bufD  = (float*)(ws + off);    off += ND4;   // agg, later h1
    float*    sden  = (float*)(ws + off);    off += NH4;   // softmax denominators
    _Float16* nf16  = (_Float16*)(ws + off); off += (size_t)NN * 32 * 2;
    _Float16* w16   = (_Float16*)(ws + off);

    _Float16* encW16 = w16;                      // 128*32 (K padded to 32)
    _Float16* wq16   = encW16 + 128 * 32;        // 3*128*128
    _Float16* wk16   = wq16 + LL * DD * DD;
    _Float16* wv16   = wk16 + LL * DD * DD;
    _Float16* wo16   = wv16 + LL * DD * DD;
    _Float16* w1_16  = wo16 + LL * DD * DD;      // 3*512*128
    _Float16* w2_16  = w1_16 + LL * DFFN * DD;   // 3*128*512

    float*    q32   = (float*)bufA;
    float*    k32   = (float*)(bufA + ND4);
    _Float16* mid16 = (_Float16*)bufA;
    float*    v32   = bufB;

    const int T = 256;
    auto cdiv = [](int a, int b) { return (a + b - 1) / b; };

    // ---- weight / input precision conversion ----
    gt_padcast_f16<<<cdiv(DD * 32, T), T, 0, stream>>>(enc_W, encW16, DD);
    gt_padcast_f16<<<cdiv(NN * 32, T), T, 0, stream>>>(nf, nf16, NN);
    gt_cast_f16<<<cdiv(LL * DD * DD, T), T, 0, stream>>>(Wq, wq16, LL * DD * DD);
    gt_cast_f16<<<cdiv(LL * DD * DD, T), T, 0, stream>>>(Wk, wk16, LL * DD * DD);
    gt_cast_f16<<<cdiv(LL * DD * DD, T), T, 0, stream>>>(Wv, wv16, LL * DD * DD);
    gt_cast_f16<<<cdiv(LL * DD * DD, T), T, 0, stream>>>(Wo, wo16, LL * DD * DD);
    gt_cast_f16<<<cdiv(LL * DFFN * DD, T), T, 0, stream>>>(W1, w1_16, LL * DFFN * DD);
    gt_cast_f16<<<cdiv(LL * DD * DFFN, T), T, 0, stream>>>(W2, w2_16, LL * DD * DFFN);

    const int MT = NN / 16;                      // 3125 exact
    dim3 wb2(32, 2);                             // Nout = 128 -> 2 waves/block
    dim3 wb8(32, 8);                             // Nout = 512 -> 8 waves/block

    // ---- encoder: h = nf @ enc_W^T + enc_b  (K padded 16->32) ----
    gt_gemm_wmma<<<dim3(MT, 1), wb2, 0, stream>>>(nf16, encW16, enc_b, h32, h16,
                                                  NN, DD, 32, 0);

    const int EG = cdiv(EE * HH, T);
    for (int l = 0; l < LL; ++l) {
        const _Float16* wq_l = wq16 + (size_t)l * DD * DD;
        const _Float16* wk_l = wk16 + (size_t)l * DD * DD;
        const _Float16* wv_l = wv16 + (size_t)l * DD * DD;
        const _Float16* wo_l = wo16 + (size_t)l * DD * DD;
        const _Float16* w1_l = w1_16 + (size_t)l * DFFN * DD;
        const _Float16* w2_l = w2_16 + (size_t)l * DD * DFFN;

        // Q, K, V projections (WMMA)
        gt_gemm_wmma<<<dim3(MT, 1), wb2, 0, stream>>>(h16, wq_l, nullptr, q32, nullptr, NN, DD, DD, 0);
        gt_gemm_wmma<<<dim3(MT, 1), wb2, 0, stream>>>(h16, wk_l, nullptr, k32, nullptr, NN, DD, DD, 0);
        gt_gemm_wmma<<<dim3(MT, 1), wb2, 0, stream>>>(h16, wv_l, nullptr, v32, nullptr, NN, DD, DD, 0);

        // edge softmax (single pass, clamp makes max-shift unnecessary)
        gt_zero_f32<<<cdiv(NN * HH, T), T, 0, stream>>>(sden, NN * HH);
        gt_edge_logits<<<EG, T, 0, stream>>>(q32, k32, src, dst, bufC, sden);

        // weighted aggregation: agg[dst] += a * v[src]
        gt_zero_f32<<<cdiv(NN * DD, T), T, 0, stream>>>(bufD, NN * DD);
        gt_edge_aggregate<<<EG, T, 0, stream>>>(v32, bufC, sden, src, dst, bufD);

        // uh = agg @ Wo^T   (agg -> f16 into h16, h16's old contents are dead)
        gt_cast_f16<<<cdiv(NN * DD, T), T, 0, stream>>>(bufD, h16, NN * DD);
        gt_gemm_wmma<<<dim3(MT, 1), wb2, 0, stream>>>(h16, wo_l, nullptr, bufC, nullptr, NN, DD, DD, 0);

        // h1 = LN(h + uh) -> bufD (f32) + h16 (f16)
        gt_add_ln<<<cdiv(NN, 8), wb8, 0, stream>>>(h32, bufC, ln1g + l * DD, ln1b + l * DD,
                                                   bufD, h16, NN);

        // FFN: mid = relu(h1 @ W1^T + b1) (f16 only), f = mid @ W2^T + b2
        gt_gemm_wmma<<<dim3(MT, 1), wb8, 0, stream>>>(h16, w1_l, b1 + l * DFFN, nullptr, mid16,
                                                      NN, DFFN, DD, 1);
        gt_gemm_wmma<<<dim3(MT, 1), wb2, 0, stream>>>(mid16, w2_l, b2 + l * DD, bufB, nullptr,
                                                      NN, DD, DFFN, 0);

        // h = LN(h1 + f) -> h32 + h16 for next layer
        gt_add_ln<<<cdiv(NN, 8), wb8, 0, stream>>>(bufD, bufB, ln2g + l * DD, ln2b + l * DD,
                                                   h32, h16, NN);
    }

    // ---- decoder + mean readout ----
    gt_out_init<<<1, 1, 0, stream>>>(out, decb);
    gt_decode_mean<<<cdiv(NN, T), T, 0, stream>>>(h32, decW, out);
}
